// RockTS_68341519614837
// MI455X (gfx1250) — matrix operand order repaired and checked by reading, not compiled, not run on hardware
//
#include <hip/hip_runtime.h>
#include <hip/hip_bf16.h>
#include <math.h>

typedef _Float16 half_t;
typedef _Float16 v8h  __attribute__((ext_vector_type(8)));
typedef _Float16 v16h __attribute__((ext_vector_type(16)));
typedef float    v8f  __attribute__((ext_vector_type(8)));

#define DMODEL 768
#define DFF    3072
#define NHEAD  12
#define LSEQ   64
#define NSEQ   448            // 64 * 7
#define ROWS   28672          // NSEQ * LSEQ
#define NELEM  22020096UL     // ROWS * DMODEL
#define ATT_SCALE 0.125f      // 64^-0.5
#define BN_EPS 1e-5f

static __device__ __forceinline__ v16h cat16(v8h lo, v8h hi) {
  v16h r;
#pragma unroll
  for (int i = 0; i < 8; ++i) { r[i] = lo[i]; r[i + 8] = hi[i]; }
  return r;
}

// ------------------------------------------- f32 [K,N] -> f16 transposed [N,K]
// 32x32 LDS tile transpose, coalesced reads and writes; blockIdx.z = layer
__global__ __launch_bounds__(256) void cvt_transpose_kernel(
    const float* __restrict__ src, half_t* __restrict__ dst, int K, int N) {
  __shared__ float tile[32][33];
  size_t loff = (size_t)blockIdx.z * K * N;
  const float* S = src + loff;
  half_t* D = dst + loff;
  int k0 = blockIdx.y * 32, n0 = blockIdx.x * 32;
  int t = threadIdx.x;
#pragma unroll
  for (int i = 0; i < 4; ++i) {
    int idx = t + i * 256;
    int r = idx >> 5, c = idx & 31;
    tile[r][c] = S[(size_t)(k0 + r) * N + n0 + c];
  }
  __syncthreads();
#pragma unroll
  for (int i = 0; i < 4; ++i) {
    int idx = t + i * 256;
    int r = idx >> 5, c = idx & 31;
    D[(size_t)(n0 + r) * K + k0 + c] = (half_t)tile[c][r];
  }
}

// ------------------------------------------------------------- patch embed
// one block per (b, n, v) row of x [BS, NP, NV, PL]
__global__ __launch_bounds__(256) void embed_kernel(
    const float* __restrict__ x, const float* __restrict__ Wp,
    const float* __restrict__ bp, const float* __restrict__ Wpos,
    float* __restrict__ u32, half_t* __restrict__ u16,
    float* __restrict__ hT) {
  __shared__ float xr[16];
  int row = blockIdx.x;              // (b*64 + n)*7 + v
  int b   = row / 448;
  int rem = row % 448;
  int n   = rem / 7;
  int vv  = rem % 7;
  int t = threadIdx.x;
  if (t < 16) xr[t] = x[(size_t)row * 16 + t];
  __syncthreads();
  size_t bvrow = (size_t)(b * 7 + vv);
#pragma unroll
  for (int rep = 0; rep < 3; ++rep) {
    int d = t + rep * 256;
    float acc = bp[d];
#pragma unroll
    for (int p = 0; p < 16; ++p) acc += xr[p] * Wp[p * DMODEL + d];
    hT[bvrow * DMODEL * LSEQ + (size_t)d * LSEQ + n] = acc;   // h (pre-pos)
    float uu = acc + Wpos[n * DMODEL + d];
    size_t uidx = (bvrow * LSEQ + n) * DMODEL + d;
    u32[uidx] = uu;
    u16[uidx] = (half_t)uu;
  }
}

// --------------------------------------------------------------- WMMA GEMM
// C[M,N] = A[M,K] (f16, row-major) @ BT[N,K] (f16, pre-transposed) + bias
// block tile 128x128x32, 8 waves (4x2 WMMA tiles each), double-buffered LDS:
// global loads for tile k+1 are issued before the WMMAs of tile k.
#define BM 128
#define BN 128
#define BKK 32
#define LDT 40   // halfs per LDS row (32 + 8 pad)

template <bool WRITE_F32, bool GELU>
__global__ __launch_bounds__(256) void gemm_wmma_kernel(
    const half_t* __restrict__ A, const half_t* __restrict__ BT,
    const float* __restrict__ bias, float* __restrict__ Cf,
    half_t* __restrict__ Ch, int N, int K) {
  __shared__ half_t As[2][BM * LDT];
  __shared__ half_t Bs[2][BN * LDT];   // Bs[n][k]
  const int tid  = threadIdx.x;
  const int wave = tid >> 5;
  const int lane = tid & 31;
  const int wm = wave >> 2;         // 0..1 (64 rows each)
  const int wn = wave & 3;          // 0..3 (32 cols each)
  const int r  = lane & 15;
  const int hh = lane >> 4;
  const int m0 = blockIdx.y * BM;
  const int n0 = blockIdx.x * BN;

  // staging coordinates: 2 chunks per thread, identical pattern for A and BT
  int srow[2], soff[2];
#pragma unroll
  for (int i = 0; i < 2; ++i) {
    int c = tid + i * 256;          // 0..511
    srow[i] = c >> 2;               // 0..127
    soff[i] = (c & 3) << 3;         // 0,8,16,24
  }

  v8h aReg[2], bReg[2];
  auto gload = [&](int k0) {
#pragma unroll
    for (int i = 0; i < 2; ++i) {
      aReg[i] = *(const v8h*)(A  + (size_t)(m0 + srow[i]) * K + k0 + soff[i]);
      bReg[i] = *(const v8h*)(BT + (size_t)(n0 + srow[i]) * K + k0 + soff[i]);
    }
  };
  auto sstore = [&](int buf) {
#pragma unroll
    for (int i = 0; i < 2; ++i) {
      *(v8h*)(&As[buf][srow[i] * LDT + soff[i]]) = aReg[i];
      *(v8h*)(&Bs[buf][srow[i] * LDT + soff[i]]) = bReg[i];
    }
  };

  const v8f vzero = {0.f, 0.f, 0.f, 0.f, 0.f, 0.f, 0.f, 0.f};
  v8f acc[4][2];
#pragma unroll
  for (int i = 0; i < 4; ++i)
#pragma unroll
    for (int j = 0; j < 2; ++j) acc[i][j] = vzero;

  gload(0);
  sstore(0);
  const int T = K / BKK;
  for (int kt = 0; kt < T; ++kt) {
    __syncthreads();
    if (kt + 1 < T) gload((kt + 1) * BKK);   // overlap with WMMAs below

    const half_t* Ab = &As[kt & 1][0];
    const half_t* Bb = &Bs[kt & 1][0];
    // A fragment: lane half 0 -> K {0..7,16..23}; half 1 -> K {8..15,24..31}
    v16h af[4];
#pragma unroll
    for (int i = 0; i < 4; ++i) {
      const half_t* p = Ab + (wm * 64 + i * 16 + r) * LDT + (hh << 3);
      af[i] = cat16(*(const v8h*)p, *(const v8h*)(p + 16));
    }
    // B fragment: lane half 0 -> K {0..7,16..23}; half 1 -> K {8..15,24..31}
    v16h bf[2];
#pragma unroll
    for (int j = 0; j < 2; ++j) {
      const half_t* p = Bb + (wn * 32 + j * 16 + r) * LDT + (hh << 3);
      bf[j] = cat16(*(const v8h*)p, *(const v8h*)(p + 16));
    }
#pragma unroll
    for (int i = 0; i < 4; ++i)
#pragma unroll
      for (int j = 0; j < 2; ++j)
        acc[i][j] = __builtin_amdgcn_wmma_f32_16x16x32_f16(
            false, af[i], false, bf[j], (short)0, acc[i][j], false, false);

    if (kt + 1 < T) sstore((kt + 1) & 1);    // other buffer: no race
  }

  // epilogue: D layout VGPR v -> M = v + 8*laneHalf, N = lane&15
#pragma unroll
  for (int j = 0; j < 2; ++j) {
    int col = n0 + wn * 32 + j * 16 + r;
    float bb = bias[col];
#pragma unroll
    for (int i = 0; i < 4; ++i) {
#pragma unroll
      for (int vv = 0; vv < 8; ++vv) {
        int row = m0 + wm * 64 + i * 16 + (hh << 3) + vv;
        float val = acc[i][j][vv] + bb;
        if (GELU) val = 0.5f * val * (1.f + erff(val * 0.70710678118f));
        size_t idx = (size_t)row * N + col;
        if (WRITE_F32) Cf[idx] = val;
        else           Ch[idx] = (half_t)val;
      }
    }
  }
}

// ---------------------------------------------------------------- attention
// one block per (seq, head): scores = QK^T * scale (+ prev raw), softmax, PV
#define QSTR 68   // half stride
#define SSTR 65   // float stride
__global__ __launch_bounds__(256) void attn_kernel(
    const half_t* __restrict__ q, const half_t* __restrict__ k,
    const half_t* __restrict__ v, float* __restrict__ scores,
    float* __restrict__ probs_out, half_t* __restrict__ o, int has_prev) {
  __shared__ half_t Qh[LSEQ * QSTR];
  __shared__ half_t Kh[LSEQ * QSTR];
  __shared__ half_t Vh[LSEQ * QSTR];
  __shared__ float  Ss[LSEQ * SSTR];
  int blk  = blockIdx.x;
  int seq  = blk / NHEAD;
  int head = blk % NHEAD;
  size_t base = (size_t)seq * LSEQ * DMODEL + head * 64;
  int t = threadIdx.x;
  for (int idx = t; idx < 4096; idx += 256) {
    int i = idx >> 6, d = idx & 63;
    size_t g = base + (size_t)i * DMODEL + d;
    Qh[i * QSTR + d] = q[g];
    Kh[i * QSTR + d] = k[g];
    Vh[i * QSTR + d] = v[g];
  }
  __syncthreads();

  size_t sbase = (size_t)blk * 4096;
  int i  = t >> 2;
  int j0 = (t & 3) << 4;
#pragma unroll 1
  for (int jj = 0; jj < 16; ++jj) {
    int j = j0 + jj;
    float acc = 0.f;
#pragma unroll 8
    for (int d = 0; d < 64; ++d)
      acc += (float)Qh[i * QSTR + d] * (float)Kh[j * QSTR + d];
    float s = acc * ATT_SCALE;
    if (has_prev) s += scores[sbase + i * 64 + j];
    scores[sbase + i * 64 + j] = s;   // raw scores carried to next layer
    Ss[i * SSTR + j] = s;
  }
  __syncthreads();
  if (t < 64) {
    float mx = -1e30f;
    for (int j = 0; j < 64; ++j) mx = fmaxf(mx, Ss[t * SSTR + j]);
    float sum = 0.f;
    for (int j = 0; j < 64; ++j) {
      float e = __expf(Ss[t * SSTR + j] - mx);
      Ss[t * SSTR + j] = e;
      sum += e;
    }
    float inv = 1.f / sum;
    for (int j = 0; j < 64; ++j) Ss[t * SSTR + j] *= inv;
  }
  __syncthreads();
  if (probs_out) {
    for (int idx = t; idx < 4096; idx += 256)
      probs_out[sbase + idx] = Ss[(idx >> 6) * SSTR + (idx & 63)];
  }
  int d0 = (t & 3) << 4;
#pragma unroll 1
  for (int dd = 0; dd < 16; ++dd) {
    int d = d0 + dd;
    float acc = 0.f;
#pragma unroll 8
    for (int j = 0; j < 64; ++j)
      acc += Ss[i * SSTR + j] * (float)Vh[j * QSTR + d];
    o[base + (size_t)i * DMODEL + d] = (half_t)acc;
  }
}

// ---------------------------------------------------------------- BatchNorm
__global__ void bn_zero_kernel(float* __restrict__ sums) {
  int i = blockIdx.x * 256 + threadIdx.x;
  if (i < 2 * DMODEL) sums[i] = 0.f;
}

// 112 blocks x 256 rows; thread owns channels t, t+256, t+512
__global__ __launch_bounds__(256) void bn_stats_kernel(
    const float* __restrict__ u, const float* __restrict__ delta,
    float* __restrict__ sums) {
  int t = threadIdx.x;
  float s0 = 0.f, s1 = 0.f, s2 = 0.f, q0 = 0.f, q1 = 0.f, q2 = 0.f;
  size_t rbase = (size_t)blockIdx.x * 256;
  for (int r = 0; r < 256; ++r) {
    size_t base = (rbase + r) * DMODEL;
    float a0 = u[base + t]       + delta[base + t];
    float a1 = u[base + t + 256] + delta[base + t + 256];
    float a2 = u[base + t + 512] + delta[base + t + 512];
    s0 += a0; q0 += a0 * a0;
    s1 += a1; q1 += a1 * a1;
    s2 += a2; q2 += a2 * a2;
  }
  atomicAdd(&sums[t],                 s0);
  atomicAdd(&sums[t + 256],           s1);
  atomicAdd(&sums[t + 512],           s2);
  atomicAdd(&sums[DMODEL + t],        q0);
  atomicAdd(&sums[DMODEL + t + 256],  q1);
  atomicAdd(&sums[DMODEL + t + 512],  q2);
}

__global__ void bn_apply_kernel(float* __restrict__ u,
                                const float* __restrict__ delta,
                                const float* __restrict__ sums,
                                const float* __restrict__ g,
                                const float* __restrict__ be,
                                half_t* __restrict__ u16) {
  const float invN = 1.f / (float)ROWS;
  for (size_t idx = (size_t)blockIdx.x * 256 + threadIdx.x; idx < NELEM;
       idx += (size_t)gridDim.x * 256) {
    int c = (int)(idx % DMODEL);
    float m = sums[c] * invN;
    float var = sums[DMODEL + c] * invN - m * m;
    float s = u[idx] + delta[idx];
    float y = (s - m) * rsqrtf(var + BN_EPS) * g[c] + be[c];
    u[idx]  = y;
    u16[idx] = (half_t)y;
  }
}

// ------------------------------------------------------------- final output
__global__ void zout_kernel(const float* __restrict__ u32,
                            float* __restrict__ z) {
  for (size_t idx = (size_t)blockIdx.x * 256 + threadIdx.x; idx < NELEM;
       idx += (size_t)gridDim.x * 256) {
    int n = (int)(idx & 63);
    size_t rest = idx >> 6;
    int d = (int)(rest % DMODEL);
    size_t bv = rest / DMODEL;
    z[idx] = u32[(bv * LSEQ + n) * DMODEL + d];
  }
}

// ============================================================== launch
extern "C" void kernel_launch(void* const* d_in, const int* in_sizes, int n_in,
                              void* d_out, int out_size, void* d_ws,
                              size_t ws_size, hipStream_t stream) {
  (void)in_sizes; (void)n_in; (void)out_size; (void)ws_size;
  const float* x    = (const float*)d_in[0];
  const float* Wp   = (const float*)d_in[1];
  const float* bp   = (const float*)d_in[2];
  const float* Wpos = (const float*)d_in[3];
  const float* Wq   = (const float*)d_in[4];
  const float* bq   = (const float*)d_in[5];
  const float* Wk   = (const float*)d_in[6];
  const float* bk   = (const float*)d_in[7];
  const float* Wv   = (const float*)d_in[8];
  const float* bv   = (const float*)d_in[9];
  const float* Wo   = (const float*)d_in[10];
  const float* bo   = (const float*)d_in[11];
  const float* g1   = (const float*)d_in[12];
  const float* be1  = (const float*)d_in[13];
  const float* W1   = (const float*)d_in[14];
  const float* b1   = (const float*)d_in[15];
  const float* W2   = (const float*)d_in[16];
  const float* b2   = (const float*)d_in[17];
  const float* g2   = (const float*)d_in[18];
  const float* be2  = (const float*)d_in[19];

  float* zout  = (float*)d_out;
  float* hTout = zout + NELEM;
  float* attn_probs = zout + 2 * NELEM;

  char* ws = (char*)d_ws;
  size_t off = 0;
  auto wsalloc = [&](size_t bytes) -> void* {
    void* p = (void*)(ws + off);
    off += (bytes + 255) & ~(size_t)255;
    return p;
  };
  float*  u32   = (float*) wsalloc((size_t)ROWS * DMODEL * 4);
  half_t* u16   = (half_t*)wsalloc((size_t)ROWS * DMODEL * 2);
  half_t* q16   = (half_t*)wsalloc((size_t)ROWS * DMODEL * 2);
  half_t* k16   = (half_t*)wsalloc((size_t)ROWS * DMODEL * 2);
  half_t* v16   = (half_t*)wsalloc((size_t)ROWS * DMODEL * 2);
  half_t* o16   = (half_t*)wsalloc((size_t)ROWS * DMODEL * 2);
  half_t* hid16 = (half_t*)wsalloc((size_t)ROWS * DFF * 2);
  float*  tmp32 = (float*) wsalloc((size_t)ROWS * DMODEL * 4);
  float*  scoresB = (float*)wsalloc((size_t)NSEQ * NHEAD * LSEQ * LSEQ * 4);
  float*  bnsum = (float*) wsalloc(2 * DMODEL * 4);
  half_t* wq16  = (half_t*)wsalloc((size_t)3 * DMODEL * DMODEL * 2);
  half_t* wk16  = (half_t*)wsalloc((size_t)3 * DMODEL * DMODEL * 2);
  half_t* wv16  = (half_t*)wsalloc((size_t)3 * DMODEL * DMODEL * 2);
  half_t* wo16  = (half_t*)wsalloc((size_t)3 * DMODEL * DMODEL * 2);
  half_t* w116  = (half_t*)wsalloc((size_t)3 * DMODEL * DFF * 2);
  half_t* w216  = (half_t*)wsalloc((size_t)3 * DFF * DMODEL * 2);

  // weights converted to f16 AND transposed to [N,K] once per launch
  dim3 gtD(DMODEL / 32, DMODEL / 32, 3);
  cvt_transpose_kernel<<<gtD, 256, 0, stream>>>(Wq, wq16, DMODEL, DMODEL);
  cvt_transpose_kernel<<<gtD, 256, 0, stream>>>(Wk, wk16, DMODEL, DMODEL);
  cvt_transpose_kernel<<<gtD, 256, 0, stream>>>(Wv, wv16, DMODEL, DMODEL);
  cvt_transpose_kernel<<<gtD, 256, 0, stream>>>(Wo, wo16, DMODEL, DMODEL);
  dim3 gt1(DFF / 32, DMODEL / 32, 3);   // W1: [DM, DFF] -> [DFF, DM]
  cvt_transpose_kernel<<<gt1, 256, 0, stream>>>(W1, w116, DMODEL, DFF);
  dim3 gt2(DMODEL / 32, DFF / 32, 3);   // W2: [DFF, DM] -> [DM, DFF]
  cvt_transpose_kernel<<<gt2, 256, 0, stream>>>(W2, w216, DFF, DMODEL);

  embed_kernel<<<ROWS, 256, 0, stream>>>(x, Wp, bp, Wpos, u32, u16, hTout);

  dim3 gD(DMODEL / BN, ROWS / BM);   // 6 x 224
  dim3 gF(DFF / BN,    ROWS / BM);   // 24 x 224

  for (int l = 0; l < 3; ++l) {
    const half_t* WQl = wq16 + (size_t)l * DMODEL * DMODEL;
    const half_t* WKl = wk16 + (size_t)l * DMODEL * DMODEL;
    const half_t* WVl = wv16 + (size_t)l * DMODEL * DMODEL;
    const half_t* WOl = wo16 + (size_t)l * DMODEL * DMODEL;
    const half_t* W1l = w116 + (size_t)l * DMODEL * DFF;
    const half_t* W2l = w216 + (size_t)l * DFF * DMODEL;

    gemm_wmma_kernel<false, false><<<gD, 256, 0, stream>>>(
        u16, WQl, bq + l * DMODEL, nullptr, q16, DMODEL, DMODEL);
    gemm_wmma_kernel<false, false><<<gD, 256, 0, stream>>>(
        u16, WKl, bk + l * DMODEL, nullptr, k16, DMODEL, DMODEL);
    gemm_wmma_kernel<false, false><<<gD, 256, 0, stream>>>(
        u16, WVl, bv + l * DMODEL, nullptr, v16, DMODEL, DMODEL);

    attn_kernel<<<NSEQ * NHEAD, 256, 0, stream>>>(
        q16, k16, v16, scoresB, (l == 2) ? attn_probs : nullptr, o16,
        (l > 0) ? 1 : 0);

    gemm_wmma_kernel<true, false><<<gD, 256, 0, stream>>>(
        o16, WOl, bo + l * DMODEL, tmp32, nullptr, DMODEL, DMODEL);

    bn_zero_kernel<<<6, 256, 0, stream>>>(bnsum);
    bn_stats_kernel<<<ROWS / 256, 256, 0, stream>>>(u32, tmp32, bnsum);
    bn_apply_kernel<<<4096, 256, 0, stream>>>(u32, tmp32, bnsum,
        g1 + l * DMODEL, be1 + l * DMODEL, u16);

    gemm_wmma_kernel<false, true><<<gF, 256, 0, stream>>>(
        u16, W1l, b1 + l * DFF, nullptr, hid16, DFF, DMODEL);
    gemm_wmma_kernel<true, false><<<gD, 256, 0, stream>>>(
        hid16, W2l, b2 + l * DMODEL, tmp32, nullptr, DMODEL, DFF);

    bn_zero_kernel<<<6, 256, 0, stream>>>(bnsum);
    bn_stats_kernel<<<ROWS / 256, 256, 0, stream>>>(u32, tmp32, bnsum);
    bn_apply_kernel<<<4096, 256, 0, stream>>>(u32, tmp32, bnsum,
        g2 + l * DMODEL, be2 + l * DMODEL, u16);
  }

  zout_kernel<<<4096, 256, 0, stream>>>(u32, zout);
}
